// LinformerAttention_45500883534106
// MI455X (gfx1250) — matrix operand order repaired and checked by reading, not compiled, not run on hardware
//
#include <hip/hip_runtime.h>
#include <hip/hip_bf16.h>
#include <stdint.h>

// ---------------------------------------------------------------------------
// Linformer attention on MI455X (gfx1250, wave32).
// All GEMMs run on v_wmma_f32_16x16x32_bf16 (fp32 accumulate).
// Tile staging uses the CDNA5 async global->LDS pipe (ASYNCcnt) with
// double-buffered LDS so copy overlaps matrix work.
// ---------------------------------------------------------------------------

typedef __attribute__((ext_vector_type(16))) __bf16          v16bf;
typedef __attribute__((ext_vector_type(8)))  __bf16          v8bf;
typedef __attribute__((ext_vector_type(8)))  float           v8f;
typedef __attribute__((ext_vector_type(8)))  unsigned short  v8u16;

__device__ __forceinline__ unsigned short f2bf(float f) {
    union { float f; unsigned u; } c; c.f = f;
    unsigned u = c.u;
    unsigned r = u + 0x7FFFu + ((u >> 16) & 1u);   // round-to-nearest-even
    return (unsigned short)(r >> 16);
}
__device__ __forceinline__ float bf2f(unsigned short h) {
    union { unsigned u; float f; } c; c.u = ((unsigned)h) << 16;
    return c.f;
}

// gfx1250 async global->LDS copy, 16B per lane, tracked by ASYNCcnt.
// LDS flat address truncated to 32 bits == LDS byte offset (ISA 10.2 aperture).
__device__ __forceinline__ void async_copy_b128(const unsigned short* lds_dst,
                                                const unsigned short* gsrc) {
    unsigned lds_addr = (unsigned)(uintptr_t)lds_dst;
    unsigned long long gaddr = (unsigned long long)(uintptr_t)gsrc;
    asm volatile("global_load_async_to_lds_b128 %0, %1, off"
                 :: "v"(lds_addr), "v"(gaddr)
                 : "memory");
}
__device__ __forceinline__ void wait_async0() {
    asm volatile("s_wait_asynccnt 0x0" ::: "memory");
}

// ---------------------------------------------------------------------------
// fp32 -> bf16 conversion (grid-stride)
// ---------------------------------------------------------------------------
__global__ void cvt_f32_bf16(const float* __restrict__ in,
                             unsigned short* __restrict__ out, long long n) {
    long long i = (long long)blockIdx.x * blockDim.x + threadIdx.x;
    long long stride = (long long)gridDim.x * blockDim.x;
    for (; i < n; i += stride) out[i] = f2bf(in[i]);
}

// fp32 (R x Ccols) -> bf16 transposed (Ccols x R); 32x32 LDS tile.
__global__ __launch_bounds__(256)
void cvt_transpose_f32_bf16(const float* __restrict__ in,
                            unsigned short* __restrict__ out,
                            int R, int Ccols) {
    __shared__ float tile[32][33];
    const int r0 = blockIdx.y * 32, c0 = blockIdx.x * 32;
    const int tx = threadIdx.x, ty = threadIdx.y;   // block (32, 8)
    #pragma unroll
    for (int i = 0; i < 32; i += 8)
        tile[ty + i][tx] = in[(long long)(r0 + ty + i) * Ccols + c0 + tx];
    __syncthreads();
    #pragma unroll
    for (int i = 0; i < 32; i += 8)
        out[(long long)(c0 + ty + i) * R + r0 + tx] = f2bf(tile[tx][ty + i]);
}

// ---------------------------------------------------------------------------
// Batched bf16 WMMA GEMM:  C[z] = A[z](MxK) * B[z](KxN) (+bias)
// Block tile 128x64, K-step 32; 8 waves/block, each wave 32x32 (2x2 WMMA).
// Double-buffered LDS, async global->LDS staging.
// TRANS_B: B given as (N x K) row-major (pre-transposed weights / k_proj^T).
// All problem dims are exact multiples of the tile, so no edge guards.
// ---------------------------------------------------------------------------
template<bool OUT_BF16, bool TRANS_B, bool HAS_BIAS>
__global__ __launch_bounds__(256)
void gemm_bf16_wmma(const unsigned short* __restrict__ A,
                    const unsigned short* __restrict__ B,
                    const float* __restrict__ bias,
                    void* __restrict__ Cout,
                    int M, int N, int K,
                    int lda, int ldb, int ldc,
                    long long sAb, long long sAh,
                    long long sBb, long long sBh,
                    long long sCb, long long sCh,
                    int batch_h) {
    constexpr int BM = 128, BN = 64, BK = 32, PAD = 8, LDW = BK + PAD;
    __shared__ unsigned short As[2][BM][LDW];   // row-major, K contiguous
    __shared__ unsigned short Bs[2][BN][LDW];   // per-column K contiguous

    const int tid  = threadIdx.x;
    const int wave = tid >> 5;
    const int lane = tid & 31;
    const int wm   = wave & 3;   // 4 waves down M (4*32 = 128)
    const int wn   = wave >> 2;  // 2 waves across N (2*32 = 64)

    const int z  = blockIdx.z;
    const int zb = z / batch_h;
    const int zh = z % batch_h;
    const unsigned short* Ab = A + zb * sAb + zh * sAh;
    const unsigned short* Bb = B + zb * sBb + zh * sBh;
    const long long cbase = zb * sCb + zh * sCh;

    const int row0 = blockIdx.y * BM;
    const int col0 = blockIdx.x * BN;

    v8f zero = {0.f, 0.f, 0.f, 0.f, 0.f, 0.f, 0.f, 0.f};
    v8f acc[2][2];
    acc[0][0] = zero; acc[0][1] = zero; acc[1][0] = zero; acc[1][1] = zero;

    // Stage-in assignments: 16B per thread per copy.
    const int a_r = tid >> 2;           // 0..63 (two passes -> 128 rows)
    const int a_c = (tid & 3) * 8;      // 0,8,16,24
    const int b_n = tid >> 2;           // 0..63  (TRANS_B path)
    const int b_k = (tid & 3) * 8;

    const int half = lane >> 4;         // lane half selects K sub-range
    const int l16  = lane & 15;

    auto stage = [&](int buf, int k0) {
        #pragma unroll
        for (int p = 0; p < 2; ++p) {
            const int r = a_r + p * 64;
            async_copy_b128(&As[buf][r][a_c],
                            Ab + (long long)(row0 + r) * lda + k0 + a_c);
        }
        if (TRANS_B) {
            async_copy_b128(&Bs[buf][b_n][b_k],
                            Bb + (long long)(col0 + b_n) * ldb + k0 + b_k);
        } else {
            // one vector global load + 8-way LDS scatter (transpose into LDS)
            const int kk = tid >> 3;          // 0..31
            const int nn = (tid & 7) * 8;     // 0..56
            v8u16 tmp = *(const v8u16*)(Bb + (long long)(k0 + kk) * ldb + col0 + nn);
            #pragma unroll
            for (int e = 0; e < 8; ++e) Bs[buf][nn + e][kk] = tmp[e];
        }
    };

    stage(0, 0);
    const int nk = K / BK;
    for (int it = 0; it < nk; ++it) {
        const int cur = it & 1;
        wait_async0();        // this wave's async copies into buf `cur` landed
        __syncthreads();      // all waves' copies landed; prev compute done
        if (it + 1 < nk) stage(1 - cur, (it + 1) * BK);

        // ---- fragments (per ISA VGPR layouts) ----
        v16bf afrag[2], bfrag[2];
        #pragma unroll
        for (int i = 0; i < 2; ++i) {
            const int ar = wm * 32 + i * 16 + l16;
            // A 16-bit layout: half 0 -> K {0..7,16..23}, half 1 -> K {8..15,24..31}
            v8u16 lo = *(const v8u16*)&As[cur][ar][half * 8];
            v8u16 hi = *(const v8u16*)&As[cur][ar][16 + half * 8];
            afrag[i] = __builtin_shufflevector(
                __builtin_bit_cast(v8bf, lo), __builtin_bit_cast(v8bf, hi),
                0, 1, 2, 3, 4, 5, 6, 7, 8, 9, 10, 11, 12, 13, 14, 15);
        }
        #pragma unroll
        for (int j = 0; j < 2; ++j) {
            const int bc = wn * 32 + j * 16 + l16;
            // B 16-bit layout: half 0 -> K {0..7,16..23}, half 1 -> K {8..15,24..31}
            v8u16 lo = *(const v8u16*)&Bs[cur][bc][half * 8];
            v8u16 hi = *(const v8u16*)&Bs[cur][bc][16 + half * 8];
            bfrag[j] = __builtin_shufflevector(
                __builtin_bit_cast(v8bf, lo), __builtin_bit_cast(v8bf, hi),
                0, 1, 2, 3, 4, 5, 6, 7, 8, 9, 10, 11, 12, 13, 14, 15);
        }

        #pragma unroll
        for (int i = 0; i < 2; ++i)
            #pragma unroll
            for (int j = 0; j < 2; ++j)
                acc[i][j] = __builtin_amdgcn_wmma_f32_16x16x32_bf16(
                    false, afrag[i], false, bfrag[j],
                    (short)0, acc[i][j], false, false);
    }

    // ---- write-out: C/D layout -> lane l16 = column, VGPR r -> row r + 8*half
    #pragma unroll
    for (int i = 0; i < 2; ++i) {
        #pragma unroll
        for (int j = 0; j < 2; ++j) {
            const int gcol = col0 + wn * 32 + j * 16 + l16;
            const float bv = HAS_BIAS ? bias[gcol] : 0.f;
            #pragma unroll
            for (int r = 0; r < 8; ++r) {
                const int grow = row0 + wm * 32 + i * 16 + half * 8 + r;
                const long long idx = cbase + (long long)grow * ldc + gcol;
                const float v = acc[i][j][r] + bv;
                if (OUT_BF16) ((unsigned short*)Cout)[idx] = f2bf(v);
                else          ((float*)Cout)[idx] = v;
            }
        }
    }
}

// ---------------------------------------------------------------------------
// In-place softmax over the projected k-axis (128 wide), scale folded in.
// ---------------------------------------------------------------------------
__global__ __launch_bounds__(128)
void softmax128_bf16(unsigned short* __restrict__ att, float scale) {
    __shared__ float red[128];
    const long long row = blockIdx.x;
    const int t = threadIdx.x;
    unsigned short* p = att + row * 128;

    const float x = bf2f(p[t]) * scale;
    red[t] = x;
    __syncthreads();
    for (int s = 64; s > 0; s >>= 1) {
        if (t < s) red[t] = fmaxf(red[t], red[t + s]);
        __syncthreads();
    }
    const float m = red[0];
    __syncthreads();
    const float e = __expf(x - m);
    red[t] = e;
    __syncthreads();
    for (int s = 64; s > 0; s >>= 1) {
        if (t < s) red[t] += red[t + s];
        __syncthreads();
    }
    p[t] = f2bf(e / red[0]);
}

// ---------------------------------------------------------------------------
// Host-side orchestration
// ---------------------------------------------------------------------------
static inline size_t al256(size_t x) { return (x + 255) & ~(size_t)255; }

extern "C" void kernel_launch(void* const* d_in, const int* in_sizes, int n_in,
                              void* d_out, int out_size, void* d_ws, size_t ws_size,
                              hipStream_t stream) {
    (void)in_sizes; (void)n_in; (void)out_size; (void)ws_size;

    constexpr int Bsz = 8, T = 2048, C = 1024, H = 16, D = 64, KP = 128;
    constexpr int C3 = 3 * C;
    constexpr long long MT = (long long)Bsz * T;           // 16384 token rows

    const float* x      = (const float*)d_in[0];
    const float* W_attn = (const float*)d_in[1];
    const float* b_attn = (const float*)d_in[2];
    const float* W_proj = (const float*)d_in[3];
    const float* b_proj = (const float*)d_in[4];
    const float* E      = (const float*)d_in[5];
    const float* F      = (const float*)d_in[6];
    float* out = (float*)d_out;

    // workspace carve-up (bf16 buffers)
    char* base = (char*)d_ws;
    size_t off = 0;
    auto carve = [&](size_t elems) {
        unsigned short* p = (unsigned short*)(base + off);
        off += al256(elems * sizeof(unsigned short));
        return p;
    };
    unsigned short* x_bf   = carve(MT * C);                       // 32 MB
    unsigned short* wa_t   = carve((size_t)C * C3);               //  6 MB  (3072x1024)
    unsigned short* wp_t   = carve((size_t)C * C);                //  2 MB  (1024x1024)
    unsigned short* e_bf   = carve((size_t)H * KP * T);           //  8 MB
    unsigned short* f_bf   = carve((size_t)H * KP * T);           //  8 MB
    unsigned short* qkv_bf = carve(MT * C3);                      // 96 MB
    unsigned short* kp_bf  = carve((size_t)Bsz * H * KP * D);     //  2 MB
    unsigned short* vp_bf  = carve((size_t)Bsz * H * KP * D);     //  2 MB
    unsigned short* att_bf = carve((size_t)Bsz * H * T * KP);     // 64 MB
    unsigned short* y_bf   = carve(MT * C);                       // 32 MB

    // 1) fp32 -> bf16 conversions; weights transposed to (N x K) for TRANS_B
    cvt_f32_bf16<<<4096, 256, 0, stream>>>(x, x_bf, MT * C);
    cvt_f32_bf16<<<4096, 256, 0, stream>>>(E, e_bf, (long long)H * KP * T);
    cvt_f32_bf16<<<4096, 256, 0, stream>>>(F, f_bf, (long long)H * KP * T);
    cvt_transpose_f32_bf16<<<dim3(C3 / 32, C / 32), dim3(32, 8), 0, stream>>>(
        W_attn, wa_t, C, C3);
    cvt_transpose_f32_bf16<<<dim3(C / 32, C / 32), dim3(32, 8), 0, stream>>>(
        W_proj, wp_t, C, C);

    // 2) qkv = x @ W_attn + b_attn              [16384 x 3072]
    gemm_bf16_wmma<true, true, true><<<dim3(C3 / 64, MT / 128, 1), 256, 0, stream>>>(
        x_bf, wa_t, b_attn, qkv_bf,
        (int)MT, C3, C, C, C, C3,
        0, 0, 0, 0, 0, 0, 1);

    // 3) k_proj[b,h] = E[h] (128 x T) @ k[b,h] (T x 64)
    gemm_bf16_wmma<true, false, false><<<dim3(1, 1, Bsz * H), 256, 0, stream>>>(
        e_bf, qkv_bf + C, nullptr, kp_bf,
        KP, D, T, T, C3, D,
        0LL, (long long)KP * T,                       // A: E indexed by h only
        (long long)T * C3, (long long)D,              // B: k slice of qkv
        (long long)H * KP * D, (long long)KP * D,     // C: k_proj[b,h]
        H);

    // 4) v_proj[b,h] = F[h] @ v[b,h]
    gemm_bf16_wmma<true, false, false><<<dim3(1, 1, Bsz * H), 256, 0, stream>>>(
        f_bf, qkv_bf + 2 * C, nullptr, vp_bf,
        KP, D, T, T, C3, D,
        0LL, (long long)KP * T,
        (long long)T * C3, (long long)D,
        (long long)H * KP * D, (long long)KP * D,
        H);

    // 5) att[b,h] = q[b,h] (T x 64) @ k_proj[b,h]^T (64 x 128)   [TRANS_B]
    gemm_bf16_wmma<true, true, false><<<dim3(KP / 64, T / 128, Bsz * H), 256, 0, stream>>>(
        qkv_bf, kp_bf, nullptr, att_bf,
        T, KP, D, C3, D, KP,
        (long long)T * C3, (long long)D,              // A: q slice of qkv
        (long long)H * KP * D, (long long)KP * D,     // B: k_proj (N x K rowmaj)
        (long long)H * T * KP, (long long)T * KP,     // C: att[b,h]
        H);

    // 6) softmax over k axis, scale = 1/sqrt(D), in place on bf16
    softmax128_bf16<<<dim3((unsigned)(Bsz * H * T)), 128, 0, stream>>>(att_bf, 0.125f);

    // 7) y[b,h] = att[b,h] (T x 128) @ v_proj[b,h] (128 x 64), scattered into
    //    the merged [B,T,C] layout via per-head C strides
    gemm_bf16_wmma<true, false, false><<<dim3(1, T / 128, Bsz * H), 256, 0, stream>>>(
        att_bf, vp_bf, nullptr, y_bf,
        T, D, KP, KP, D, C,
        (long long)H * T * KP, (long long)T * KP,
        (long long)H * KP * D, (long long)KP * D,
        (long long)T * C, (long long)D,               // C: y[b, :, h*64 + ...]
        H);

    // 8) out = y @ W_proj + b_proj   (fp32 output)   [16384 x 1024]
    gemm_bf16_wmma<false, true, true><<<dim3(C / 64, MT / 128, 1), 256, 0, stream>>>(
        y_bf, wp_t, b_proj, out,
        (int)MT, C, C, C, C, C,
        0, 0, 0, 0, 0, 0, 1);
}
